// Attention_84026740179675
// MI455X (gfx1250) — matrix operand order repaired and checked by reading, not compiled, not run on hardware
//
#include <hip/hip_runtime.h>

// ---------------------------------------------------------------------------
// Types for CDNA5 WMMA (wave32, 16x16x32 bf16 -> f32)
// ---------------------------------------------------------------------------
typedef __attribute__((ext_vector_type(16))) __bf16       v16bf;
typedef __attribute__((ext_vector_type(8)))  float        v8f;
typedef __attribute__((ext_vector_type(4)))  unsigned int u32x4;

union AF {                       // one 16x32 bf16 A/B fragment (8 VGPRs/lane)
    v16bf          v;
    u32x4          q[2];
    unsigned short h[16];
};
union Pack8 { unsigned short h[8]; u32x4 q; };

// Hardware bf16 convert (fptrunc -> v_cvt_*bf16_f32, RNE)
__device__ __forceinline__ unsigned short f32_to_bf16(float f) {
    __bf16 b = (__bf16)f;
    return __builtin_bit_cast(unsigned short, b);
}

// Native V_TANH_F32 if the toolchain exposes it; ocml fallback otherwise.
#if __has_builtin(__builtin_amdgcn_tanhf)
#define FAST_TANH(x) __builtin_amdgcn_tanhf(x)
#else
#define FAST_TANH(x) tanhf(x)
#endif

#define WMMA_BF16(A, Bm, C) \
    __builtin_amdgcn_wmma_f32_16x16x32_bf16(false, (A), false, (Bm), (short)0, (C), false, false)

static constexpr int S_  = 1024;
static constexpr int B_  = 16;
static constexpr int D_  = 512;
static constexpr int H_  = 8;
static constexpr int DH_ = 64;

// ---------------------------------------------------------------------------
// fp32 -> bf16 conversion
// ---------------------------------------------------------------------------
__global__ void cvt_bf16_kernel(const float* __restrict__ src,
                                unsigned short* __restrict__ dst, int n) {
    int i = blockIdx.x * blockDim.x + threadIdx.x;
    if (i < n) dst[i] = f32_to_bf16(src[i]);
}

// ---------------------------------------------------------------------------
// QKV projection: one wave per (b, h, 16-row query tile).
// q[b,h,s,e] = sum_d x[s,b,d] * w[h,e,d]   (GEMM M=16, N=64, K=512)
// Q,K stored [B,H,S,DH] bf16; V stored transposed [B,H,DH,S] bf16.
// ---------------------------------------------------------------------------
__global__ __launch_bounds__(256) void proj_kernel(
    const unsigned short* __restrict__ xb,    // [S,B,D]  bf16
    const unsigned short* __restrict__ wqb,   // [H,DH,D] bf16
    const unsigned short* __restrict__ wkb,
    const unsigned short* __restrict__ wvb,
    unsigned short* __restrict__ qb,          // [B,H,S,DH]
    unsigned short* __restrict__ kb,          // [B,H,S,DH]
    unsigned short* __restrict__ vt)          // [B,H,DH,S]
{
    const int lane  = threadIdx.x & 31;
    const int wave  = (blockIdx.x * blockDim.x + threadIdx.x) >> 5; // 0..8191
    const int stile = wave & 63;          // S/16 tiles
    const int bh    = wave >> 6;          // 0..127
    const int h     = bh & (H_ - 1);
    const int b     = bh >> 3;
    const int s0    = stile * 16;
    const int lhalf = lane >> 4;          // 0 | 1
    const int lmod  = lane & 15;

    v8f accq[4] = {}, acck[4] = {}, accv[4] = {};

    for (int k0 = 0; k0 < D_; k0 += 32) {
        // A fragment: x rows s0..s0+15, cols k0..k0+31  (doc 16-bit A layout)
        AF a;
        {
            const unsigned short* row = xb + ((s0 + lmod) * B_ + b) * D_ + k0 + lhalf * 8;
            a.q[0] = *(const u32x4*)(row);        // K = lhalf*8 .. +7
            a.q[1] = *(const u32x4*)(row + 16);   // K = 16+lhalf*8 .. +7
        }
        auto dot = [&](const unsigned short* w, v8f* acc) {
#pragma unroll
            for (int n = 0; n < 4; ++n) {
                AF bf;
                const unsigned short* col = w + (h * DH_ + n * 16 + lmod) * D_
                                              + k0 + lhalf * 8;
                bf.q[0] = *(const u32x4*)(col);       // K = lhalf*8 .. +7
                bf.q[1] = *(const u32x4*)(col + 16);  // K = 16+lhalf*8 .. +7
                acc[n]  = WMMA_BF16(a.v, bf.v, acc[n]);
            }
        };
        dot(wqb, accq);
        dot(wkb, acck);
        dot(wvb, accv);
    }

    // Store. C layout: VGPR r, lane -> (M = r + 8*lhalf, N = lmod)
    const int base_qk = (b * H_ + h) * S_ * DH_;
#pragma unroll
    for (int n = 0; n < 4; ++n) {
        Pack8 pk;
#pragma unroll
        for (int r = 0; r < 8; ++r) {
            const int s = s0 + r + 8 * lhalf;
            const int e = n * 16 + lmod;
            qb[base_qk + s * DH_ + e] = f32_to_bf16(accq[n][r]);
            kb[base_qk + s * DH_ + e] = f32_to_bf16(acck[n][r]);
            pk.h[r]                   = f32_to_bf16(accv[n][r]);
        }
        // vt[b,h,e,s]: 8 consecutive s per lane -> one 16B store
        const int e = n * 16 + lmod;
        *(u32x4*)(vt + ((b * H_ + h) * DH_ + e) * S_ + s0 + 8 * lhalf) = pk.q;
    }
}

// ---------------------------------------------------------------------------
// Flash-style attention: one wave per (b, h, 16-query tile).
// tanh bounds logits by 1 -> fixed softmax shift c=1, no max tracking.
// ---------------------------------------------------------------------------
__global__ __launch_bounds__(256) void attn_kernel(
    const unsigned short* __restrict__ qb,   // [B,H,S,DH]
    const unsigned short* __restrict__ kb,   // [B,H,S,DH]
    const unsigned short* __restrict__ vt,   // [B,H,DH,S]
    const float* __restrict__ mask,          // [B,S]
    float* __restrict__ out)                 // [S,B,D]
{
    __shared__ unsigned short pbuf[8][16 * 32];   // per-wave P staging (8 KB)

    const int wid   = threadIdx.x >> 5;
    const int lane  = threadIdx.x & 31;
    const int wave  = blockIdx.x * 8 + wid;
    const int stile = wave & 63;
    const int bh    = wave >> 6;
    const int h     = bh & (H_ - 1);
    const int b     = bh >> 3;
    const int s0    = stile * 16;
    const int lhalf = lane >> 4;
    const int lmod  = lane & 15;

    // Q A-fragments (dh 0..31 and 32..63), loaded once
    AF qa[2];
    {
        const unsigned short* qrow =
            qb + ((b * H_ + h) * S_ + s0 + lmod) * DH_ + lhalf * 8;
        qa[0].q[0] = *(const u32x4*)(qrow);
        qa[0].q[1] = *(const u32x4*)(qrow + 16);
        qa[1].q[0] = *(const u32x4*)(qrow + 32);
        qa[1].q[1] = *(const u32x4*)(qrow + 48);
    }

    v8f   o[4]    = {};
    float lsum[8] = {};
    const float inv_sqrt_dh = 0.125f;   // 1/sqrt(64)

    unsigned short*       pp    = &pbuf[wid][0];
    const unsigned short* kbase = kb + (b * H_ + h) * S_ * DH_;
    const unsigned short* vbase = vt + (b * H_ + h) * DH_ * S_;
    const float*          mrow  = mask + b * S_;

    for (int kb0 = 0; kb0 < S_; kb0 += 32) {
        // ---- scores: 2 tiles of (16 q x 16 keys), K-dim = DH = 2x32 ----
        v8f sc[2] = {};
#pragma unroll
        for (int t = 0; t < 2; ++t) {
#pragma unroll
            for (int j = 0; j < 2; ++j) {
                AF kf;
                const unsigned short* kr =
                    kbase + (kb0 + t * 16 + lmod) * DH_ + j * 32 + lhalf * 8;
                kf.q[0] = *(const u32x4*)(kr);
                kf.q[1] = *(const u32x4*)(kr + 16);
                sc[t]   = WMMA_BF16(qa[j].v, kf.v, sc[t]);
            }
        }

        // ---- p = mask * exp(tanh(s/8) - 1); running row sums; stage to LDS
#pragma unroll
        for (int t = 0; t < 2; ++t) {
            const float mv = mrow[kb0 + t * 16 + lmod];
#pragma unroll
            for (int r = 0; r < 8; ++r) {
                const float p = mv * __expf(FAST_TANH(sc[t][r] * inv_sqrt_dh) - 1.0f);
                lsum[r] += p;
                pp[(r + 8 * lhalf) * 32 + t * 16 + lmod] = f32_to_bf16(p);
            }
        }
        asm volatile("s_wait_dscnt 0x0" ::: "memory");   // LDS stores visible

        // ---- read P back as a 16x32 A fragment ----
        AF pa;
        {
            const unsigned short* prow = pp + lmod * 32 + lhalf * 8;
            pa.q[0] = *(const u32x4*)(prow);
            pa.q[1] = *(const u32x4*)(prow + 16);
        }

        // ---- O += P @ V  (B-frag contiguous along keys thanks to V^T) ----
#pragma unroll
        for (int n = 0; n < 4; ++n) {
            AF vf;
            const unsigned short* vr =
                vbase + (n * 16 + lmod) * S_ + kb0 + lhalf * 8;
            vf.q[0] = *(const u32x4*)(vr);
            vf.q[1] = *(const u32x4*)(vr + 16);
            o[n]    = WMMA_BF16(pa.v, vf.v, o[n]);
        }
    }

    // ---- reduce row sums across the 16 lanes holding the same rows ----
#pragma unroll
    for (int r = 0; r < 8; ++r) {
        float v = lsum[r];
        v += __shfl_xor(v, 1, 32);
        v += __shfl_xor(v, 2, 32);
        v += __shfl_xor(v, 4, 32);
        v += __shfl_xor(v, 8, 32);
        lsum[r] = v;
    }

    // ---- normalize and store out[s, b, h*64 + e] ----
#pragma unroll
    for (int n = 0; n < 4; ++n) {
#pragma unroll
        for (int r = 0; r < 8; ++r) {
            const int s = s0 + r + 8 * lhalf;
            const int e = h * DH_ + n * 16 + lmod;
            out[(s * B_ + b) * D_ + e] = o[n][r] / lsum[r];
        }
    }
}

// ---------------------------------------------------------------------------
// Host launcher
// ---------------------------------------------------------------------------
extern "C" void kernel_launch(void* const* d_in, const int* in_sizes, int n_in,
                              void* d_out, int out_size, void* d_ws, size_t ws_size,
                              hipStream_t stream) {
    (void)in_sizes; (void)n_in; (void)out_size; (void)ws_size;
    const float* x    = (const float*)d_in[0];
    const float* mask = (const float*)d_in[1];
    const float* wq   = (const float*)d_in[2];
    const float* wk   = (const float*)d_in[3];
    const float* wv   = (const float*)d_in[4];
    float*       out  = (float*)d_out;

    const int XE = S_ * B_ * D_;        // 8388608
    const int WE = H_ * DH_ * D_;       // 262144
    const int QE = B_ * H_ * S_ * DH_;  // 8388608

    unsigned short* ws  = (unsigned short*)d_ws;
    unsigned short* xb  = ws;
    unsigned short* wqb = xb  + XE;
    unsigned short* wkb = wqb + WE;
    unsigned short* wvb = wkb + WE;
    unsigned short* qbp = wvb + WE;
    unsigned short* kbp = qbp + QE;
    unsigned short* vtp = kbp + QE;

    cvt_bf16_kernel<<<(XE + 255) / 256, 256, 0, stream>>>(x,  xb,  XE);
    cvt_bf16_kernel<<<(WE + 255) / 256, 256, 0, stream>>>(wq, wqb, WE);
    cvt_bf16_kernel<<<(WE + 255) / 256, 256, 0, stream>>>(wk, wkb, WE);
    cvt_bf16_kernel<<<(WE + 255) / 256, 256, 0, stream>>>(wv, wvb, WE);

    // 8192 waves each = 1024 blocks x 8 waves
    proj_kernel<<<1024, 256, 0, stream>>>(xb, wqb, wkb, wvb, qbp, kbp, vtp);
    attn_kernel<<<1024, 256, 0, stream>>>(qbp, kbp, vtp, mask, out);
}
